// GRU_39170101739852
// MI455X (gfx1250) — matrix operand order repaired and checked by reading, not compiled, not run on hardware
//
#include <hip/hip_runtime.h>

// CDNA5 / gfx1250 GRU: wave32, WMMA f16 16x16x32.
// Workgroup = 4 waves cooperating on ONE 16-row batch tile: wave g owns the
// r/z/n gate tiles for hidden columns 16g..16g+15 (6 WMMAs + 8-element gate
// math per step), recurrence closed via double-buffered LDS h + 1 barrier/step.
// 256 blocks x 4 waves = 1024 waves of parallelism over B=4096.

typedef __attribute__((ext_vector_type(16))) _Float16 v16h;
typedef __attribute__((ext_vector_type(8)))  _Float16 v8h;
typedef __attribute__((ext_vector_type(8)))  float    v8f;
typedef __attribute__((ext_vector_type(4)))  float    v4f;

#define B_TOT 4096
#define T_LEN 512
#define H_DIM 64
#define TCH   32     // time chunk for x/out staging

union V16 { v16h v; v8h h[2]; };

__device__ __forceinline__ float fast_rcp(float a) {
  return __builtin_amdgcn_rcpf(a);            // v_rcp_f32
}
__device__ __forceinline__ float fast_sigmoid(float x) {
  return fast_rcp(1.0f + __expf(-x));         // v_exp_f32 + v_rcp_f32
}
__device__ __forceinline__ float fast_tanh(float x) {
  return 1.0f - 2.0f * fast_rcp(1.0f + __expf(2.0f * x));
}

__global__ __launch_bounds__(128) void gru_wmma_kernel(
    const float* __restrict__ x,      // [B, T] (flattened [B,T,1])
    const float* __restrict__ w_ih,   // [192]  ([3H,1])
    const float* __restrict__ w_hh,   // [192, 64]
    const float* __restrict__ b_ih,   // [192]
    const float* __restrict__ b_hh,   // [192]
    const float* __restrict__ w_lin,  // [64]   ([1,H])
    const float* __restrict__ b_lin,  // [1]
    float* __restrict__ out)          // [B, T]
{
  // w_hh^T staged as 24 f16 WMMA-B tiles (gate-tile gt 0..11, K-tile kt 0..1),
  // stored in per-consumer-lane operand order: (tile*32 + lane)*16 halves.
  __shared__ __align__(16) _Float16 wstage[24 * 32 * 16];   // 24 KB
  __shared__ __align__(16) _Float16 hstage[2][16 * H_DIM];  // 4 KB (double buffer)
  __shared__ __align__(16) float    xstage[TCH * 16];       // 2 KB
  __shared__ __align__(16) float    ostage[16 * TCH];       // 2 KB

  const int tid   = threadIdx.x;
  const int lane  = tid & 31;
  const int wave  = tid >> 5;    // gate group g = wave (hidden cols 16g..16g+15)
  const int col   = lane & 15;   // N-column within a 16x16 tile
  const int lh    = lane >> 4;   // lane half (0/1)
  const int batchBase = blockIdx.x * 16;

  // ---- stage w_hh^T -> LDS f16 B-tiles (cooperative, once) ----
  // Consumer lane l of tile (gt,kt): n = l%16 (gate col), kblock = 32*kt + 16*(l/16),
  // halves i=0..15 hold w_hh^T[kblock+i][16*gt+n] = w_hh[(16*gt+n)*64 + kblock + i].
  for (int slot = tid; slot < 24 * 32; slot += 128) {
    const int tile = slot >> 5, l = slot & 31;
    const int gt = tile >> 1, kt = tile & 1;
    const int gate = 16 * gt + (l & 15);
    const int kb   = 32 * kt + 16 * (l >> 4);
    const float* src = w_hh + gate * H_DIM + kb;
    _Float16* dst = wstage + slot * 16;
    #pragma unroll
    for (int i = 0; i < 16; ++i) dst[i] = (_Float16)src[i];
  }
  __syncthreads();

  // ---- per-lane time-invariant gate constants for this wave's gate group ----
  const int gc = 16 * wave + col;                  // gate column (r section)
  const float wihc_r = w_ih[gc];
  const float wihc_z = w_ih[64 + gc];
  const float wihc_n = w_ih[128 + gc];
  const float bsum_r = b_ih[gc]      + b_hh[gc];
  const float bsum_z = b_ih[64 + gc] + b_hh[64 + gc];
  const float bihn   = b_ih[128 + gc];
  const float bhhn   = b_hh[128 + gc];
  const float blin   = b_lin[0];

  // ---- register-resident w_lin as WMMA-B tiles (only N-col 0 non-zero) ----
  V16 bw[2];
  #pragma unroll
  for (int kt = 0; kt < 2; ++kt) {
    const int kb = 32 * kt + 16 * lh;
    #pragma unroll
    for (int i = 0; i < 16; ++i)
      bw[kt].v[i] = (col == 0) ? (_Float16)w_lin[kb + i] : (_Float16)0.0f;
  }

  // this wave's B-tile base: tile id = gt*2+kt, gt in {wave, wave+4, wave+8}
  // offsets from base (halves): r kt: +kt*512 ; z: +4096 ; n: +8192
  const _Float16* wbase = wstage + (wave * 2 * 32 + lane) * 16;

  // ---- recurrent state ----
  V16 a0, a1;                   // full h as WMMA-A f16 tiles (K = 0..31 / 32..63)
  #pragma unroll
  for (int i = 0; i < 16; ++i) { a0.v[i] = (_Float16)0.0f; a1.v[i] = (_Float16)0.0f; }
  v8f hprev = {0,0,0,0,0,0,0,0};  // this wave's h quarter, C/D layout

  for (int t = 0; t < T_LEN; ++t) {
    const int tmod = t & (TCH - 1);

    if (tmod == 0) {
      if (wave == 0) {
        // stage x[batchBase+m][t .. t+31] -> xstage[tl][m], coalesced per lane
        const int m = col, tb = lh * 16;
        const float* xp = x + (size_t)(batchBase + m) * T_LEN + t + tb;
        #pragma unroll
        for (int j = 0; j < 4; ++j) {
          v4f xv = *(const v4f*)(xp + 4 * j);
          #pragma unroll
          for (int i = 0; i < 4; ++i) xstage[(tb + 4 * j + i) * 16 + m] = xv[i];
        }
      }
      __syncthreads();           // chunk staging visible to all waves
    }

    // per-VGPR x broadcast: xb[v] = x[batch = v + 8*lh][t]
    v4f xlo = *(const v4f*)&xstage[tmod * 16 + 8 * lh];
    v4f xhi = *(const v4f*)&xstage[tmod * 16 + 8 * lh + 4];
    float xb[8] = {xlo[0], xlo[1], xlo[2], xlo[3], xhi[0], xhi[1], xhi[2], xhi[3]};

    // ---- gh tiles for this wave's gate group: 6 WMMAs ----
    v8f accr = {0,0,0,0,0,0,0,0};
    v8f accz = {0,0,0,0,0,0,0,0};
    v8f accn = {0,0,0,0,0,0,0,0};
    #pragma unroll
    for (int kt = 0; kt < 2; ++kt) {
      V16 br, bz, bn;
      const _Float16* p = wbase + kt * 512;
      br.h[0] = *(const v8h*)p;            br.h[1] = *(const v8h*)(p + 8);
      bz.h[0] = *(const v8h*)(p + 4096);   bz.h[1] = *(const v8h*)(p + 4104);
      bn.h[0] = *(const v8h*)(p + 8192);   bn.h[1] = *(const v8h*)(p + 8200);
      v16h a = (kt == 0) ? a0.v : a1.v;
      accr = __builtin_amdgcn_wmma_f32_16x16x32_f16(false, a, false, br.v, (short)0, accr, false, false);
      accz = __builtin_amdgcn_wmma_f32_16x16x32_f16(false, a, false, bz.v, (short)0, accz, false, false);
      accn = __builtin_amdgcn_wmma_f32_16x16x32_f16(false, a, false, bn.v, (short)0, accn, false, false);
    }

    // ---- gate math on this wave's 8 elements (C/D layout) ----
    _Float16* hb = hstage[t & 1];
    v8f hnew;
    #pragma unroll
    for (int v = 0; v < 8; ++v) {
      float pr = accr[v] + xb[v] * wihc_r + bsum_r;
      float pz = accz[v] + xb[v] * wihc_z + bsum_z;
      float r  = fast_sigmoid(pr);
      float z  = fast_sigmoid(pz);
      float pn = xb[v] * wihc_n + bihn + r * (accn[v] + bhhn);
      float n  = fast_tanh(pn);
      float h  = n + z * (hprev[v] - n);
      hnew[v] = h;
      hb[(v + 8 * lh) * H_DIM + 16 * wave + col] = (_Float16)h;  // ds_store_b16
    }
    hprev = hnew;

    __syncthreads();             // h_t complete across all 4 waves

    // ---- reload full h_t as WMMA-A tiles: lane m = col, K-subblock = 16*lh ----
    {
      const _Float16* hr = hstage[t & 1];
      const int m = col, kb = 16 * lh;
      a0.h[0] = *(const v8h*)&hr[m * H_DIM +      kb];
      a0.h[1] = *(const v8h*)&hr[m * H_DIM + 8 + kb];
      a1.h[0] = *(const v8h*)&hr[m * H_DIM + 32 + kb];
      a1.h[1] = *(const v8h*)&hr[m * H_DIM + 40 + kb];
    }

    // ---- wave 0: output projection via WMMA + chunked coalesced store ----
    if (wave == 0) {
      v8f dout = { blin, blin, blin, blin, blin, blin, blin, blin };
      dout = __builtin_amdgcn_wmma_f32_16x16x32_f16(false, a0.v, false, bw[0].v, (short)0, dout, false, false);
      dout = __builtin_amdgcn_wmma_f32_16x16x32_f16(false, a1.v, false, bw[1].v, (short)0, dout, false, false);
      if (col == 0) {
        #pragma unroll
        for (int v = 0; v < 8; ++v)
          ostage[(v + 8 * lh) * TCH + tmod] = dout[v];
      }
      if (tmod == TCH - 1) {
        asm volatile("s_wait_dscnt 0" ::: "memory");
        const int m = col, tb = lh * 16;
        float* op = out + (size_t)(batchBase + m) * T_LEN + (t - TCH + 1) + tb;
        #pragma unroll
        for (int j = 0; j < 4; ++j) {
          v4f ov = *(const v4f*)&ostage[m * TCH + tb + 4 * j];
          *(v4f*)(op + 4 * j) = ov;   // global_store_b128
        }
      }
    }
  }
}

extern "C" void kernel_launch(void* const* d_in, const int* in_sizes, int n_in,
                              void* d_out, int out_size, void* d_ws, size_t ws_size,
                              hipStream_t stream) {
  (void)in_sizes; (void)n_in; (void)d_ws; (void)ws_size; (void)out_size;
  const float* x     = (const float*)d_in[0];
  const float* w_ih  = (const float*)d_in[1];
  const float* w_hh  = (const float*)d_in[2];
  const float* b_ih  = (const float*)d_in[3];
  const float* b_hh  = (const float*)d_in[4];
  const float* w_lin = (const float*)d_in[5];
  const float* b_lin = (const float*)d_in[6];
  float* out = (float*)d_out;

  dim3 grid(B_TOT / 16);   // 256 blocks, each = 4 waves sharing one 16-row batch tile
  dim3 block(128);
  gru_wmma_kernel<<<grid, block, 0, stream>>>(x, w_ih, w_hh, b_ih, b_hh, w_lin, b_lin, out);
}
